// RLNeighborSelector_62388694942428
// MI455X (gfx1250) — matrix operand order repaired and checked by reading, not compiled, not run on hardware
//
#include <hip/hip_runtime.h>
#include <hip/hip_bf16.h>

// ---------------------------------------------------------------------------
// RLNeighborSelector for MI455X (gfx1250, wave32, WMMA).
//
// Math refactor: enc[e] = relu(P[row[e]] + Q[col[e]] + b_fold) where
//   P = x @ (node_W @ enc_W[0:128])
//   Q = x @ (nbr_W @ enc_W[128:256]) + guidance @ (gud_W @ enc_W[256:384])
//   b_fold = node_b@enc_W1 + nbr_b@enc_W2 + gud_b@enc_W3 + enc_b
// Then per-edge heads on 16-edge tiles with v_wmma_f32_16x16x32_f16:
//   value = relu(enc@vp_W1+vp_b1)@vp_W2+vp_b2
//   prob  = sigmoid(relu(enc@pol_W1+pol_b1)@pol_W2+pol_b2)
// Head weights are staged per-workgroup into LDS with CDNA5 async
// global->LDS copies (global_load_async_to_lds_b128 / s_wait_asynccnt).
// ---------------------------------------------------------------------------

typedef __attribute__((ext_vector_type(16))) _Float16 v16h;
typedef __attribute__((ext_vector_type(8)))  float    v8f;

#define NNODES 50000
#define NEDGES 800000
#define HID    128

// Workspace layout (bytes). Total required ~51.6 MB.
#define P_OFF      0u
#define Q_OFF      25600000u          // NNODES*HID*4
#define BFOLD_OFF  51200000u
#define WFOLD_OFF  51200512u          // 3 * 128*128 f32 fused weights
#define PACK_OFF   51397120u          // packed f16 B-matrices (contiguous!)
// packed halfs: Wp 16384, Wq 16384, Wg 16384, vp1 8192, pol1 16384
// NOTE: vp1 pack and pol1 pack are contiguous (48KB) -> one LDS staging copy.

// ---------------------------------------------------------------------------
// Kernel 1: fold the three input-side weight matrices through enc_W, and
// fold all first-stage biases into one 128-vector.
// ---------------------------------------------------------------------------
__global__ __launch_bounds__(256) void fold_weights_kernel(
    const float* __restrict__ node_W, const float* __restrict__ nbr_W,
    const float* __restrict__ gud_W,  const float* __restrict__ enc_W,
    const float* __restrict__ node_b, const float* __restrict__ nbr_b,
    const float* __restrict__ gud_b,  const float* __restrict__ enc_b,
    float* __restrict__ Wfold, float* __restrict__ bfold) {
  int tid = blockIdx.x * blockDim.x + threadIdx.x;
  if (tid < 3 * HID * HID) {
    int m   = tid / (HID * HID);
    int rem = tid % (HID * HID);
    int a   = rem / HID;
    int c   = rem % HID;
    const float* Wi = (m == 0) ? node_W : (m == 1) ? nbr_W : gud_W;
    float s = 0.f;
    #pragma unroll 8
    for (int b = 0; b < HID; ++b)
      s += Wi[a * HID + b] * enc_W[(m * HID + b) * HID + c];
    Wfold[tid] = s;
  } else if (tid < 3 * HID * HID + HID) {
    int c = tid - 3 * HID * HID;
    float s = enc_b[c];
    for (int b = 0; b < HID; ++b) {
      s += node_b[b] * enc_W[b * HID + c];
      s += nbr_b[b]  * enc_W[(HID + b) * HID + c];
      s += gud_b[b]  * enc_W[(2 * HID + b) * HID + c];
    }
    bfold[c] = s;
  }
}

// ---------------------------------------------------------------------------
// Kernel 2: pack a [K x N] f32 row-major weight matrix into the wave32 WMMA
// B-matrix f16 layout: for col-group cg, k-tile t (32 K values), lane L holds
// 16 contiguous halfs: B[t*32 + (L>=16?8:0) + (i<8 ? i : 8+i)][cg*16 + (L&15)].
// Reader fetches one v16h (32B) per lane per WMMA.
// ---------------------------------------------------------------------------
__global__ __launch_bounds__(256) void pack_b_kernel(
    const float* __restrict__ W, _Float16* __restrict__ out, int K, int N) {
  int tid = blockIdx.x * blockDim.x + threadIdx.x;
  if (tid >= K * N) return;
  int i    = tid & 15;
  int lane = (tid >> 4) & 31;
  int rest = tid >> 9;
  int KT   = K >> 5;
  int t    = rest % KT;
  int cg   = rest / KT;
  int k = t * 32 + ((lane >> 4) << 3) + (i < 8 ? i : 8 + i);
  int n = cg * 16 + (lane & 15);
  out[tid] = (_Float16)W[k * N + n];
}

// ---------------------------------------------------------------------------
// Kernel 3: node-level precompute of P and Q via v_wmma_f32_16x16x32_f16.
// One wave per (16-node tile, which in {P,Q}); A tiles held in VGPRs and
// reused across all 8 output column groups.
// ---------------------------------------------------------------------------
__device__ __forceinline__ void load_a_tiles(const float* __restrict__ src,
                                             int node, int khalf, v16h a[4]) {
  #pragma unroll
  for (int t = 0; t < 4; ++t) {
    const float* p = src + node * HID + t * 32 + khalf * 8;
    #pragma unroll
    for (int i = 0; i < 8; ++i) a[t][i]     = (_Float16)p[i];
    #pragma unroll
    for (int i = 0; i < 8; ++i) a[t][8 + i] = (_Float16)p[16 + i];
  }
}

__global__ __launch_bounds__(256) void node_precompute_kernel(
    const float* __restrict__ x, const float* __restrict__ gfeat,
    const _Float16* __restrict__ WpP, const _Float16* __restrict__ WqP,
    const _Float16* __restrict__ WgP,
    float* __restrict__ P, float* __restrict__ Q) {
  int gwave = (blockIdx.x * blockDim.x + threadIdx.x) >> 5;
  int lane  = threadIdx.x & 31;
  int which = gwave & 1;          // 0 -> P, 1 -> Q
  int tile  = gwave >> 1;
  if (tile >= NNODES / 16) return;

  int M = lane & 15, khalf = lane >> 4;
  int node = tile * 16 + M;

  v16h a1[4];
  load_a_tiles(x, node, khalf, a1);
  v16h a2[4];
  if (which) load_a_tiles(gfeat, node, khalf, a2);

  const v16h* B1 = (const v16h*)(which == 0 ? WpP : WqP);
  const v16h* B2 = (const v16h*)WgP;
  float* Out = which == 0 ? P : Q;

  for (int cg = 0; cg < 8; ++cg) {
    v8f c = {};
    #pragma unroll
    for (int t = 0; t < 4; ++t)
      c = __builtin_amdgcn_wmma_f32_16x16x32_f16(
          false, a1[t], false, B1[(cg * 4 + t) * 32 + lane],
          (short)0, c, false, false);
    if (which) {
      #pragma unroll
      for (int t = 0; t < 4; ++t)
        c = __builtin_amdgcn_wmma_f32_16x16x32_f16(
            false, a2[t], false, B2[(cg * 4 + t) * 32 + lane],
            (short)0, c, false, false);
    }
    #pragma unroll
    for (int j = 0; j < 8; ++j) {
      int row = tile * 16 + j + (khalf << 3);
      Out[row * HID + cg * 16 + (lane & 15)] = c[j];
    }
  }
}

// ---------------------------------------------------------------------------
// CDNA5 async global->LDS copy (tracked by ASYNCcnt).
// ---------------------------------------------------------------------------
__device__ __forceinline__ void async_g2lds_b128(unsigned lds_off,
                                                 unsigned long long gaddr) {
  asm volatile("global_load_async_to_lds_b128 %0, %1, off"
               :: "v"(lds_off), "v"(gaddr) : "memory");
}
__device__ __forceinline__ void wait_asynccnt0() {
  asm volatile("s_wait_asynccnt 0x0" ::: "memory");
}

// ---------------------------------------------------------------------------
// Kernel 4: edge stage. One wave per 16 edges. Lane L builds the WMMA
// A-operand for edge row M=L&15 directly from global: the two half-lanes
// (khalf 0/1) jointly cover each 128-feature row disjointly, so the gather
// + b_fold + ReLU + f16-convert happens straight into A registers (no LDS
// transpose needed). Head B-matrices (48KB) are async-staged into LDS once
// per workgroup and shared by all 8 waves: 16 WMMAs (value head) + 32 WMMAs
// (policy head) per wave, then xor-shuffle reductions for the final dots.
// ---------------------------------------------------------------------------
__global__ __launch_bounds__(256) void edge_kernel(
    const long long* __restrict__ edge_index,
    const float* __restrict__ P, const float* __restrict__ Q,
    const float* __restrict__ bfold,
    const _Float16* __restrict__ headP,   // vp1 pack (8192 h) ++ pol1 pack (16384 h)
    const float* __restrict__ vp_b1, const float* __restrict__ vp_W2,
    const float* __restrict__ vp_b2,
    const float* __restrict__ pol_b1, const float* __restrict__ pol_W2,
    const float* __restrict__ pol_b2,
    float* __restrict__ out) {
  __shared__ __align__(32) _Float16 ldsB[24576];  // 48KB: vp1 [0,8192) pol1 [8192,24576)

  int tid  = threadIdx.x;
  int wave = tid >> 5;
  int lane = tid & 31;

  // --- async-stage 48KB of packed head weights into LDS (once per block) ---
  {
    unsigned ldsBase = (unsigned)(unsigned long long)(void*)&ldsB[0];
    unsigned long long gBase = (unsigned long long)headP;
    #pragma unroll
    for (int i = 0; i < 12; ++i) {
      unsigned byteOff = (unsigned)(i * 256 + tid) * 16u;
      async_g2lds_b128(ldsBase + byteOff, gBase + byteOff);
    }
    wait_asynccnt0();
  }
  __syncthreads();

  int gwave = blockIdx.x * 8 + wave;       // grid exactly covers 50000 waves
  int e0    = gwave * 16;
  int M = lane & 15, khalf = lane >> 4;

  // --- per-lane gather + fuse directly into WMMA A layout ---
  int r = (int)edge_index[e0 + M];
  int c = (int)edge_index[NEDGES + e0 + M];
  const float* pr = P + r * HID;
  const float* qc = Q + c * HID;
  v16h a[4];
  #pragma unroll
  for (int t = 0; t < 4; ++t) {
    int k0 = t * 32 + khalf * 8;
    #pragma unroll
    for (int i = 0; i < 8; ++i) {
      float v = pr[k0 + i] + qc[k0 + i] + bfold[k0 + i];
      a[t][i] = (_Float16)(v > 0.f ? v : 0.f);
    }
    #pragma unroll
    for (int i = 0; i < 8; ++i) {
      float v = pr[k0 + 16 + i] + qc[k0 + 16 + i] + bfold[k0 + 16 + i];
      a[t][8 + i] = (_Float16)(v > 0.f ? v : 0.f);
    }
  }

  const v16h* Bv = (const v16h*)&ldsB[0];
  const v16h* Bp = (const v16h*)&ldsB[8192];

  // --- value head: enc @ vp_W1 (128x64), relu, dot vp_W2 ---
  float accv[8];
  #pragma unroll
  for (int j = 0; j < 8; ++j) accv[j] = 0.f;
  for (int cg = 0; cg < 4; ++cg) {
    v8f cacc = {};
    #pragma unroll
    for (int t = 0; t < 4; ++t)
      cacc = __builtin_amdgcn_wmma_f32_16x16x32_f16(
          false, a[t], false, Bv[(cg * 4 + t) * 32 + lane],
          (short)0, cacc, false, false);
    float b1 = vp_b1[cg * 16 + M];
    float w2 = vp_W2[cg * 16 + M];
    #pragma unroll
    for (int j = 0; j < 8; ++j) {
      float h = cacc[j] + b1;
      accv[j] += (h > 0.f ? h : 0.f) * w2;
    }
  }

  // --- policy head: enc @ pol_W1 (128x128), relu, dot pol_W2 ---
  float accp[8];
  #pragma unroll
  for (int j = 0; j < 8; ++j) accp[j] = 0.f;
  for (int cg = 0; cg < 8; ++cg) {
    v8f cacc = {};
    #pragma unroll
    for (int t = 0; t < 4; ++t)
      cacc = __builtin_amdgcn_wmma_f32_16x16x32_f16(
          false, a[t], false, Bp[(cg * 4 + t) * 32 + lane],
          (short)0, cacc, false, false);
    float b1 = pol_b1[cg * 16 + M];
    float w2 = pol_W2[cg * 16 + M];
    #pragma unroll
    for (int j = 0; j < 8; ++j) {
      float h = cacc[j] + b1;
      accp[j] += (h > 0.f ? h : 0.f) * w2;
    }
  }

  // --- reduce across the 16 lanes of each half (wave32 xor shuffles) ---
  #pragma unroll
  for (int j = 0; j < 8; ++j) {
    float v = accv[j], p = accp[j];
    #pragma unroll
    for (int m = 1; m < 16; m <<= 1) {
      v += __shfl_xor(v, m, 32);
      p += __shfl_xor(p, m, 32);
    }
    accv[j] = v;
    accp[j] = p;
  }

  if (M == 0) {
    float b2v = vp_b2[0], b2p = pol_b2[0];
    #pragma unroll
    for (int j = 0; j < 8; ++j) {
      int e = e0 + j + (khalf << 3);
      float val   = accv[j] + b2v;
      float logit = accp[j] + b2p;
      float prob  = 1.f / (1.f + __expf(-logit));
      out[e]              = val;
      out[NEDGES + e]     = prob;
      out[2 * NEDGES + e] = prob > 0.5f ? 1.f : 0.f;
    }
  }
  if (blockIdx.x == 0 && threadIdx.x == 0) out[3 * NEDGES] = 0.f;  // value_loss
}

// ---------------------------------------------------------------------------
extern "C" void kernel_launch(void* const* d_in, const int* in_sizes, int n_in,
                              void* d_out, int out_size, void* d_ws, size_t ws_size,
                              hipStream_t stream) {
  const float*     x          = (const float*)d_in[0];
  const long long* edge_index = (const long long*)d_in[1];
  const float*     gfeat      = (const float*)d_in[2];
  const float*     node_W     = (const float*)d_in[3];
  const float*     node_b     = (const float*)d_in[4];
  const float*     nbr_W      = (const float*)d_in[5];
  const float*     nbr_b      = (const float*)d_in[6];
  const float*     gud_W      = (const float*)d_in[7];
  const float*     gud_b      = (const float*)d_in[8];
  const float*     enc_W      = (const float*)d_in[9];
  const float*     enc_b      = (const float*)d_in[10];
  const float*     vp_W1      = (const float*)d_in[11];
  const float*     vp_b1      = (const float*)d_in[12];
  const float*     vp_W2      = (const float*)d_in[13];
  const float*     vp_b2      = (const float*)d_in[14];
  const float*     pol_W1     = (const float*)d_in[15];
  const float*     pol_b1     = (const float*)d_in[16];
  const float*     pol_W2     = (const float*)d_in[17];
  const float*     pol_b2     = (const float*)d_in[18];

  char* ws = (char*)d_ws;
  float*     P     = (float*)(ws + P_OFF);
  float*     Q     = (float*)(ws + Q_OFF);
  float*     bfold = (float*)(ws + BFOLD_OFF);
  float*     Wfold = (float*)(ws + WFOLD_OFF);
  _Float16*  WpP   = (_Float16*)(ws + PACK_OFF);
  _Float16*  WqP   = WpP  + 16384;
  _Float16*  WgP   = WqP  + 16384;
  _Float16*  vp1P  = WgP  + 16384;   // head packs start here (contiguous 48KB)
  _Float16*  pol1P = vp1P + 8192;

  // 1) fold weights + biases (tiny)
  fold_weights_kernel<<<(3 * HID * HID + HID + 255) / 256, 256, 0, stream>>>(
      node_W, nbr_W, gud_W, enc_W, node_b, nbr_b, gud_b, enc_b, Wfold, bfold);

  // 2) pack all B matrices into f16 WMMA layout
  pack_b_kernel<<<64, 256, 0, stream>>>(Wfold,                 WpP,   HID, HID);
  pack_b_kernel<<<64, 256, 0, stream>>>(Wfold + HID * HID,     WqP,   HID, HID);
  pack_b_kernel<<<64, 256, 0, stream>>>(Wfold + 2 * HID * HID, WgP,   HID, HID);
  pack_b_kernel<<<32, 256, 0, stream>>>(vp_W1,                 vp1P,  HID, 64);
  pack_b_kernel<<<64, 256, 0, stream>>>(pol_W1,                pol1P, HID, HID);

  // 3) node-level P/Q precompute: 3125 tiles x {P,Q} = 6250 waves
  node_precompute_kernel<<<(6250 + 7) / 8, 256, 0, stream>>>(
      x, gfeat, WpP, WqP, WgP, P, Q);

  // 4) edge stage: 50000 waves (16 edges each), 8 waves per block
  edge_kernel<<<6250, 256, 0, stream>>>(
      edge_index, P, Q, bfold, vp1P,
      vp_b1, vp_W2, vp_b2, pol_b1, pol_W2, pol_b2, (float*)d_out);
}
